// AiterRMSNormWithAddModel_83734682402839
// MI455X (gfx1250) — compile-verified, matrix operand order checked by reading, not run on hardware
//
#include <hip/hip_runtime.h>
#include <hip/hip_bf16.h>

// ---------------------------------------------------------------------------
// AiterRMSNormWithAddModel: relu -> rmsnorm -> 3x (GEMM -> +resid -> rmsnorm)
// T=32768, H=2048, fp32 in/out. GEMMs use v_wmma_f32_16x16x32_bf16 with LDS
// double-buffering fed by global_load_async_to_lds_b128 (ASYNCcnt pipeline).
// Residual stream and rmsnorm statistics stay fp32.
// ---------------------------------------------------------------------------

#define T_TOK 32768
#define HID   2048
#define KC    32          // K-chunk staged in LDS per pipeline stage
#define MT    256         // block M tile (8 waves x 32 rows)
#define NT    64          // block N tile
#define LDS_S 40          // padded row stride (ushorts): 20 dwords -> conflict-free

typedef __attribute__((ext_vector_type(16))) __bf16          v16bf;
typedef __attribute__((ext_vector_type(8)))  float           v8f;
typedef __attribute__((ext_vector_type(8)))  unsigned short  u16x8;
typedef __attribute__((ext_vector_type(4)))  unsigned int    u32x4;

union BFrag {
    v16bf bf;
    u16x8 h[2];
};

#if defined(__HIP_DEVICE_COMPILE__) && __has_builtin(__builtin_amdgcn_global_load_async_to_lds_b128)
#define ASYNC_PATH 1
#else
#define ASYNC_PATH 0
#endif

#if ASYNC_PATH
// Builtin param types (from clang diag): int4-vector pointers, AS1 src / AS3 dst.
typedef int v4i_vs __attribute__((vector_size(4 * sizeof(int))));
typedef __attribute__((address_space(1))) v4i_vs as1_v4i;
typedef __attribute__((address_space(3))) v4i_vs as3_v4i;
#endif

// 16-byte global -> LDS copy (async on CDNA5, sync fallback otherwise)
static __device__ __forceinline__ void cp16_g2l(const unsigned short* g, unsigned short* l) {
#if ASYNC_PATH
    __builtin_amdgcn_global_load_async_to_lds_b128((as1_v4i*)g, (as3_v4i*)l, 0, 0);
#else
    *(u32x4*)l = *(const u32x4*)g;
#endif
}

static __device__ __forceinline__ void wait_async0() {
#if ASYNC_PATH
#if __has_builtin(__builtin_amdgcn_s_wait_asynccnt)
    __builtin_amdgcn_s_wait_asynccnt(0);
#else
    asm volatile("s_wait_asynccnt 0" ::: "memory");
#endif
#endif
}

static __device__ __forceinline__ unsigned short f32_to_bf16_rne(float f) {
    union { float f; unsigned int u; } v;
    v.f = f;
    unsigned int lsb = (v.u >> 16) & 1u;
    unsigned int r = v.u + 0x7fffu + lsb;
    return (unsigned short)(r >> 16);
}

// ---------------------------------------------------------------------------
// Weight convert + transpose: W[K,N] fp32 -> Wt[N,K] bf16 (LDS-tiled), so B
// fragments load with the same "16 rows x contiguous K" shape as A fragments
// (CDNA5 B layout: B rows stripe across lanes => lane n holds column n of B).
// ---------------------------------------------------------------------------
__global__ __launch_bounds__(256) void cvt_w_kernel(const float* __restrict__ W,
                                                    unsigned short* __restrict__ Wt) {
    __shared__ unsigned short tile[32][33];
    const int tx = threadIdx.x;   // 0..31
    const int ty = threadIdx.y;   // 0..7
    const int n0 = blockIdx.x * 32;
    const int k0 = blockIdx.y * 32;
#pragma unroll
    for (int j = 0; j < 4; ++j) {
        const int k = k0 + ty + j * 8;
        const int n = n0 + tx;
        tile[ty + j * 8][tx] = f32_to_bf16_rne(W[(size_t)k * HID + n]);
    }
    __syncthreads();
#pragma unroll
    for (int j = 0; j < 4; ++j) {
        const int n = n0 + ty + j * 8;
        const int k = k0 + tx;
        Wt[(size_t)n * HID + k] = tile[tx][ty + j * 8];
    }
}

// ---------------------------------------------------------------------------
// Row kernel: optional relu, rmsnorm over H=2048, optional outputs.
// One 256-thread block (8 wave32) per row; 8 elements per thread.
// ---------------------------------------------------------------------------
__global__ __launch_bounds__(256) void rownorm_kernel(const float* __restrict__ in,
                                                      const float* __restrict__ gain,
                                                      float* __restrict__ resid_out,
                                                      unsigned short* __restrict__ ybf,
                                                      float* __restrict__ yf,
                                                      int do_relu) {
    __shared__ float red[8];
    const int row = blockIdx.x;
    const int t   = threadIdx.x;
    const float* p = in + (size_t)row * HID;

    float v[8];
    float ss = 0.f;
#pragma unroll
    for (int j = 0; j < 8; ++j) {
        float f = p[t + j * 256];
        if (do_relu) f = fmaxf(f, 0.f);
        v[j] = f;
        ss += f * f;
    }
#pragma unroll
    for (int off = 16; off > 0; off >>= 1)
        ss += __shfl_down(ss, off, 32);
    if ((t & 31) == 0) red[t >> 5] = ss;
    __syncthreads();
    float tot = 0.f;
#pragma unroll
    for (int w = 0; w < 8; ++w) tot += red[w];
    const float rs = rsqrtf(tot * (1.0f / (float)HID) + 1e-6f);

#pragma unroll
    for (int j = 0; j < 8; ++j) {
        const int c = t + j * 256;
        const float g = gain[c];
        const float o = v[j] * rs * g;
        if (resid_out) resid_out[(size_t)row * HID + c] = v[j];
        if (ybf)       ybf[(size_t)row * HID + c] = f32_to_bf16_rne(o);
        if (yf)        yf[(size_t)row * HID + c] = o;
    }
}

// ---------------------------------------------------------------------------
// Async stage: fill one LDS double-buffer slot with a 256x32 A tile and a
// 64x32 B tile (bf16). A: 1024 16B chunks, B: 256 chunks over 256 threads
// -> 5 async b128 loads per thread per stage.
// ---------------------------------------------------------------------------
static __device__ __forceinline__ void stage_load(const unsigned short* __restrict__ A,
                                                  const unsigned short* __restrict__ Bt,
                                                  unsigned short* lA, unsigned short* lB,
                                                  int rowBase, int colBase, int k0, int tid) {
#pragma unroll
    for (int j = 0; j < 4; ++j) {
        const int c = tid + j * 256;
        const int r = c >> 2, kc = c & 3;
        cp16_g2l(A + (size_t)(rowBase + r) * HID + k0 + kc * 8, lA + r * LDS_S + kc * 8);
    }
    {
        const int r = tid >> 2, kc = tid & 3;
        cp16_g2l(Bt + (size_t)(colBase + r) * HID + k0 + kc * 8, lB + r * LDS_S + kc * 8);
    }
}

// ---------------------------------------------------------------------------
// GEMM: resid[T,H] += A_bf16[T,H] @ W, with Bt = W^T bf16 [N,K].
// Block = 256 threads = 8 waves; block tile 256(M) x 64(N); each wave owns a
// 32x64 strip = 2 A-frags x 4 B-frags = 8 v8f accumulators.
// Pipeline per 32-wide K-chunk:
//   s_wait_asynccnt 0 -> barrier -> issue async loads for chunk k+1
//   -> 12 ds_load_b128 (all fragments) -> 8 back-to-back WMMAs.
// grid.x = N tiles (fast) so resident blocks share A panels + all of Wt in L2.
// ---------------------------------------------------------------------------
__global__ __launch_bounds__(256) void gemm_bf16_acc_kernel(const unsigned short* __restrict__ A,
                                                            const unsigned short* __restrict__ Bt,
                                                            float* __restrict__ resid) {
    __shared__ unsigned short lsA[2][MT * LDS_S];   // 2 x 20KB
    __shared__ unsigned short lsB[2][NT * LDS_S];   // 2 x 5KB

    const int tid   = threadIdx.x;
    const int wave  = tid >> 5;
    const int lane  = tid & 31;
    const int m16   = lane & 15;
    const int khalf = lane >> 4;
    const int kOff  = khalf * 8;     // element offset within the 32-wide k-step

    const int rowBase = blockIdx.y * MT;
    const int colBase = blockIdx.x * NT;

    v8f acc[8] = {};                 // [nt*2 + mf]

    stage_load(A, Bt, lsA[0], lsB[0], rowBase, colBase, 0, tid);

    int buf = 0;
    for (int k0 = 0; k0 < HID; k0 += KC) {
        wait_async0();
        __syncthreads();             // buffer `buf` filled; prev reads drained

        if (k0 + KC < HID)
            stage_load(A, Bt, lsA[buf ^ 1], lsB[buf ^ 1], rowBase, colBase, k0 + KC, tid);

        // Batch all fragment loads first, then the WMMA burst.
        BFrag a[2], b[4];
#pragma unroll
        for (int mf = 0; mf < 2; ++mf) {
            const unsigned short* a0 =
                &lsA[buf][(wave * 32 + mf * 16 + m16) * LDS_S + kOff];
            a[mf].h[0] = *(const u16x8*)(a0);
            a[mf].h[1] = *(const u16x8*)(a0 + 16);
        }
#pragma unroll
        for (int nt = 0; nt < 4; ++nt) {
            const unsigned short* b0 =
                &lsB[buf][(nt * 16 + m16) * LDS_S + kOff];
            b[nt].h[0] = *(const u16x8*)(b0);
            b[nt].h[1] = *(const u16x8*)(b0 + 16);
        }
#pragma unroll
        for (int nt = 0; nt < 4; ++nt)
#pragma unroll
            for (int mf = 0; mf < 2; ++mf)
                acc[nt * 2 + mf] = __builtin_amdgcn_wmma_f32_16x16x32_bf16(
                    false, a[mf].bf, false, b[nt].bf, (short)0,
                    acc[nt * 2 + mf], false, false);

        buf ^= 1;
    }

    // Epilogue: resid += C.  C layout: vgpr j, lanes 0-15 -> (M=j, N=lane);
    // lanes 16-31 -> (M=8+j, N=lane-16).
#pragma unroll
    for (int nt = 0; nt < 4; ++nt) {
        const int col = colBase + nt * 16 + m16;
#pragma unroll
        for (int mf = 0; mf < 2; ++mf) {
            const int rbase = rowBase + wave * 32 + mf * 16 + khalf * 8;
#pragma unroll
            for (int j = 0; j < 8; ++j) {
                float* p = resid + (size_t)(rbase + j) * HID + col;
                *p += acc[nt * 2 + mf][j];
            }
        }
    }
}

// ---------------------------------------------------------------------------
// Launch: inputs (x, g0, g1, g2, g3, W0, W1, W2), all fp32.
// Workspace: resid fp32 [T,H] | y bf16 [T,H] | Wt0/1/2 bf16 [H,H].
// ---------------------------------------------------------------------------
extern "C" void kernel_launch(void* const* d_in, const int* in_sizes, int n_in,
                              void* d_out, int out_size, void* d_ws, size_t ws_size,
                              hipStream_t stream) {
    (void)in_sizes; (void)n_in; (void)out_size; (void)ws_size;

    const float* x  = (const float*)d_in[0];
    const float* g0 = (const float*)d_in[1];
    const float* g1 = (const float*)d_in[2];
    const float* g2 = (const float*)d_in[3];
    const float* g3 = (const float*)d_in[4];
    const float* W0 = (const float*)d_in[5];
    const float* W1 = (const float*)d_in[6];
    const float* W2 = (const float*)d_in[7];
    float* out = (float*)d_out;

    float*          resid = (float*)d_ws;
    unsigned short* ybf   = (unsigned short*)(resid + (size_t)T_TOK * HID);
    unsigned short* wt0   = ybf + (size_t)T_TOK * HID;
    unsigned short* wt1   = wt0 + (size_t)HID * HID;
    unsigned short* wt2   = wt1 + (size_t)HID * HID;

    {
        dim3 grid(HID / 32, HID / 32), blk(32, 8);
        cvt_w_kernel<<<grid, blk, 0, stream>>>(W0, wt0);
        cvt_w_kernel<<<grid, blk, 0, stream>>>(W1, wt1);
        cvt_w_kernel<<<grid, blk, 0, stream>>>(W2, wt2);
    }

    // Stage 1: x1 = relu(x); resid = x1; y = rmsnorm(x1, g0) -> bf16
    rownorm_kernel<<<T_TOK, 256, 0, stream>>>(x, g0, resid, ybf, nullptr, 1);

    dim3 ggrid(HID / NT, T_TOK / MT);   // x = N tiles (fast), y = M tiles

    gemm_bf16_acc_kernel<<<ggrid, 256, 0, stream>>>(ybf, wt0, resid);
    rownorm_kernel<<<T_TOK, 256, 0, stream>>>(resid, g1, nullptr, ybf, nullptr, 0);

    gemm_bf16_acc_kernel<<<ggrid, 256, 0, stream>>>(ybf, wt1, resid);
    rownorm_kernel<<<T_TOK, 256, 0, stream>>>(resid, g2, nullptr, ybf, nullptr, 0);

    gemm_bf16_acc_kernel<<<ggrid, 256, 0, stream>>>(ybf, wt2, resid);
    rownorm_kernel<<<T_TOK, 256, 0, stream>>>(resid, g3, nullptr, nullptr, out, 0);
}
